// DeformConv_58935541236111
// MI455X (gfx1250) — compile-verified, matrix operand order checked by reading, not run on hardware
//
#include <hip/hip_runtime.h>

typedef __attribute__((ext_vector_type(16))) _Float16 v16h;
typedef __attribute__((ext_vector_type(8)))  _Float16 v8h;
typedef __attribute__((ext_vector_type(8)))  float    v8f;

union AFrag { uint4 u[2]; v16h h; };
union BFrag { unsigned int u[8]; v16h h; };
union HPack { v8h h; uint4 u; };

#define H_ 64
#define W_ 64
#define C1_ 256
#define C2_ 256
#define KK_ 9
#define KDIM 2304          // C1 * KK, ordered k = kk*256 + c
#define KSTEPS 72          // KDIM / 32 (WMMA k-chunks)
#define NSTEP2 36          // KDIM / 64 (LDS tile builds)
#define NPIX 16384         // B*H*W
#define CS64 36            // 64 halves (32 u32) + 4 u32 pad per pixel row

// ---------------- pre-pass: x (NCHW f32) -> x_h (NHWC f16) ----------------
__global__ void convert_x_kernel(const float* __restrict__ x, _Float16* __restrict__ xh) {
  int idx = blockIdx.x * 256 + threadIdx.x;       // over B*C1*H*W, NCHW order
  int xw = idx & 63; int t = idx >> 6;
  int y = t & 63; t >>= 6;
  int c = t & 255; int b = t >> 8;
  xh[(((b * H_ + y) * W_ + xw) * C1_) + c] = (_Float16)x[idx];
}

// -------- pre-pass: pack weights into wave32 16x32 A-fragment layout --------
// pack flat index = ((mtile*72 + kstep)*32 + lane)*16 + half
__device__ __forceinline__ int kmap(int lane, int h) {
  int koff = (lane < 16) ? 0 : 8;
  return (h < 8) ? (koff + h) : (16 + koff + (h - 8));
}

__global__ void pack_dcn_kernel(const float* __restrict__ w, _Float16* __restrict__ wp) {
  int idx = blockIdx.x * 256 + threadIdx.x;       // 16*72*32*16 = 589824
  int h = idx & 15; int t1 = idx >> 4;
  int lane = t1 & 31; int t2 = t1 >> 5;
  int s = t2 % KSTEPS; int mt = t2 / KSTEPS;
  int m = mt * 16 + (lane & 15);
  int k = s * 32 + kmap(lane, h);
  int kk = k >> 8; int c = k & 255;               // k = kk*256 + c
  wp[idx] = (_Float16)w[(m * C1_ + c) * KK_ + kk];
}

__global__ void pack_om_kernel(const float* __restrict__ w, _Float16* __restrict__ wp) {
  int idx = blockIdx.x * 256 + threadIdx.x;       // 2*72*32*16 = 73728
  int h = idx & 15; int t1 = idx >> 4;
  int lane = t1 & 31; int t2 = t1 >> 5;
  int s = t2 % KSTEPS; int mt = t2 / KSTEPS;
  int m = mt * 16 + (lane & 15);
  int k = s * 32 + kmap(lane, h);
  int kk = k >> 8; int c = k & 255;
  float v = (m < 27) ? w[(m * C1_ + c) * KK_ + kk] : 0.0f;
  wp[idx] = (_Float16)v;
}

// ---------------- offset/mask conv: WMMA implicit GEMM, M=32 ----------------
// block: 64 pixels; 8 waves -> 8 tiles 16x16 (waveM in {0,1}, waveN in 0..3)
// double-buffered 64K x 64pix col tile, 1 barrier per 64-K step
__global__ void om_conv_kernel(const _Float16* __restrict__ xh,
                               const uint4* __restrict__ wpk,
                               const float* __restrict__ bom,
                               float* __restrict__ om_ws) {
  __shared__ unsigned int lcol[2][64 * CS64];
  int tid = threadIdx.x;
  int lane = tid & 31, wave = tid >> 5;
  int waveM = wave & 1, waveN = wave >> 1;
  int p0 = blockIdx.x * 64;
  int b = p0 >> 12;
  int e = tid & 63, cg = tid >> 6;                // sampler role
  int hw = (p0 + e) & 4095;
  int ey = hw >> 6, ex = hw & 63;

  // prologue: build tile for s2 = 0 (kk=0, cp=0) into buffer 0
  {
    int sy = ey - 1, sx = ex - 1;                 // kk = 0 -> ky=kx=0, minus pad
    bool valid = (sy >= 0) && (sx >= 0);          // upper bounds can't trip at kk=0
    int base = ((b * H_ + (valid ? sy : 0)) * W_ + (valid ? sx : 0)) * C1_ + cg * 8;
    uint4 d0 = make_uint4(0u, 0u, 0u, 0u), d1 = d0;
    if (valid) {
      d0 = *(const uint4*)(xh + base);
      d1 = *(const uint4*)(xh + base + 32);
    }
    *(uint4*)&lcol[0][e * CS64 + cg * 4]      = d0;
    *(uint4*)&lcol[0][e * CS64 + 16 + cg * 4] = d1;
  }

  v8f acc = {};
  for (int s2 = 0; s2 < NSTEP2; ++s2) {
    __syncthreads();
    // ---- current-step A fragments (issue loads first) ----
    AFrag a[2];
#pragma unroll
    for (int ksub = 0; ksub < 2; ++ksub) {
      const uint4* ap = wpk + ((waveM * KSTEPS + (s2 * 2 + ksub)) * 32 + lane) * 2;
      a[ksub].u[0] = ap[0]; a[ksub].u[1] = ap[1];
    }
    // ---- next-tile sample loads (in flight behind WMMAs) ----
    int s2n = (s2 + 1 < NSTEP2) ? (s2 + 1) : (NSTEP2 - 1);
    int kkn = s2n >> 2, cpn = s2n & 3;
    int ky = (kkn * 11) >> 5;                     // kkn / 3 for 0..8
    int kx = kkn - 3 * ky;
    int sy = ey + ky - 1, sx = ex + kx - 1;
    bool valid = (sy >= 0) && (sy < H_) && (sx >= 0) && (sx < W_);
    int base = ((b * H_ + (valid ? sy : 0)) * W_ + (valid ? sx : 0)) * C1_ + cpn * 64 + cg * 8;
    uint4 d0 = make_uint4(0u, 0u, 0u, 0u), d1 = d0;
    if (valid) {
      d0 = *(const uint4*)(xh + base);
      d1 = *(const uint4*)(xh + base + 32);
    }
    // ---- compute current tile ----
    const unsigned int* cb = lcol[s2 & 1];
    int pix = waveN * 16 + (lane & 15);
    int ko = (lane < 16) ? 0 : 4;
#pragma unroll
    for (int ksub = 0; ksub < 2; ++ksub) {
      BFrag bf;
      int bb = pix * CS64 + ksub * 16;
#pragma unroll
      for (int j = 0; j < 4; ++j) {
        bf.u[j]     = cb[bb + ko + j];
        bf.u[4 + j] = cb[bb + 8 + ko + j];
      }
      acc = __builtin_amdgcn_wmma_f32_16x16x32_f16(false, a[ksub].h, false, bf.h,
                                                   (short)0, acc, false, false);
    }
    // ---- store next tile into other buffer ----
    unsigned int* nb = lcol[(s2 + 1) & 1];
    *(uint4*)&nb[e * CS64 + cg * 4]      = d0;
    *(uint4*)&nb[e * CS64 + 16 + cg * 4] = d1;
  }

  int n = waveN * 16 + (lane & 15);
  int p = p0 + n;
  int mbase = waveM * 16 + ((lane >= 16) ? 8 : 0);
#pragma unroll
  for (int r = 0; r < 8; ++r) {
    int m = mbase + r;
    om_ws[p * 32 + m] = acc[r] + ((m < 27) ? bom[m] : 0.0f);
  }
}

// ---------------- main fused DCN GEMM + BN + SiLU ----------------
// block tile: M=128 (blockIdx.y in {0,1}) x N=64 pixels; 8 waves, each 32x32
struct SMeta { unsigned int a00, a01, a10, a11; float w00, w01, w10, w11; };

__global__ void dcn_main_kernel(const _Float16* __restrict__ xh,
                                const uint4* __restrict__ wpk,
                                const float* __restrict__ om_ws,
                                const float* __restrict__ gamma,
                                const float* __restrict__ beta,
                                const float* __restrict__ mean,
                                const float* __restrict__ var,
                                float* __restrict__ out) {
  __shared__ SMeta smeta[KK_ * 64];
  __shared__ unsigned int lcol[2][64 * CS64];
  __shared__ float bnA[128], bnB[128];
  int tid = threadIdx.x;
  int lane = tid & 31, wave = tid >> 5;
  int waveM = wave & 3, waveN = wave >> 2;
  int p0 = blockIdx.x * 64;
  int b = p0 >> 12;

  if (tid < 128) {
    int m = blockIdx.y * 128 + tid;
    float inv = gamma[m] * rsqrtf(var[m] + 1e-5f);
    bnA[tid] = inv;
    bnB[tid] = beta[m] - mean[m] * inv;
  }
  // per-(kk,pixel) bilinear metadata (corner addrs + mask-folded weights)
  for (int i = tid; i < KK_ * 64; i += 256) {
    int kk = i >> 6, e = i & 63;
    int p = p0 + e;
    int hw = p & 4095;
    int ey = hw >> 6, ex = hw & 63;
    float oy = om_ws[p * 32 + 2 * kk];
    float ox = om_ws[p * 32 + 2 * kk + 1];
    oy = fminf(fmaxf(oy, -6.0f), 6.0f);
    ox = fminf(fmaxf(ox, -6.0f), 6.0f);
    float mk = 1.0f / (1.0f + expf(-om_ws[p * 32 + 18 + kk]));
    int ky = (kk * 11) >> 5;
    int kx = kk - 3 * ky;
    float py = (float)(ey - 1 + ky) + oy;
    float px = (float)(ex - 1 + kx) + ox;
    float fy = floorf(py), fx = floorf(px);
    int y0 = (int)fy, x0 = (int)fx;
    float ly = py - fy, lx = px - fx;
    int y1 = y0 + 1, x1 = x0 + 1;
    bool vy0 = (y0 >= 0) && (y0 < H_), vy1 = (y1 >= 0) && (y1 < H_);
    bool vx0 = (x0 >= 0) && (x0 < W_), vx1 = (x1 >= 0) && (x1 < W_);
    int y0c = min(max(y0, 0), H_ - 1), y1c = min(max(y1, 0), H_ - 1);
    int x0c = min(max(x0, 0), W_ - 1), x1c = min(max(x1, 0), W_ - 1);
    SMeta sm;
    sm.a00 = (unsigned int)(((b * H_ + y0c) * W_ + x0c) * C1_);
    sm.a01 = (unsigned int)(((b * H_ + y0c) * W_ + x1c) * C1_);
    sm.a10 = (unsigned int)(((b * H_ + y1c) * W_ + x0c) * C1_);
    sm.a11 = (unsigned int)(((b * H_ + y1c) * W_ + x1c) * C1_);
    sm.w00 = (1.0f - ly) * (1.0f - lx) * ((vy0 && vx0) ? mk : 0.0f);
    sm.w01 = (1.0f - ly) * lx * ((vy0 && vx1) ? mk : 0.0f);
    sm.w10 = ly * (1.0f - lx) * ((vy1 && vx0) ? mk : 0.0f);
    sm.w11 = ly * lx * ((vy1 && vx1) ? mk : 0.0f);
    smeta[i] = sm;
  }
  __syncthreads();

  int e = tid & 63, cg = tid >> 6;
  // prologue: build tile for s2 = 0 (kk=0, cp=0) into buffer 0
  {
    SMeta sm = smeta[e];                          // kk = 0
    int c0 = cg * 8;                              // cp = 0
    v8h v00 = *(const v8h*)(xh + sm.a00 + c0);
    v8h v01 = *(const v8h*)(xh + sm.a01 + c0);
    v8h v10 = *(const v8h*)(xh + sm.a10 + c0);
    v8h v11 = *(const v8h*)(xh + sm.a11 + c0);
    v8h u00 = *(const v8h*)(xh + sm.a00 + c0 + 32);
    v8h u01 = *(const v8h*)(xh + sm.a01 + c0 + 32);
    v8h u10 = *(const v8h*)(xh + sm.a10 + c0 + 32);
    v8h u11 = *(const v8h*)(xh + sm.a11 + c0 + 32);
    _Float16 w00 = (_Float16)sm.w00, w01 = (_Float16)sm.w01;
    _Float16 w10 = (_Float16)sm.w10, w11 = (_Float16)sm.w11;
    HPack r0; r0.h = v00 * w00 + v01 * w01 + v10 * w10 + v11 * w11;
    HPack r1; r1.h = u00 * w00 + u01 * w01 + u10 * w10 + u11 * w11;
    *(uint4*)&lcol[0][e * CS64 + cg * 4]      = r0.u;
    *(uint4*)&lcol[0][e * CS64 + 16 + cg * 4] = r1.u;
  }

  v8f acc[2][2] = {{{}, {}}, {{}, {}}};
  int mt0 = blockIdx.y * 8 + waveM * 2;
  for (int s2 = 0; s2 < NSTEP2; ++s2) {
    __syncthreads();
    // ---- current-step A fragments (issue these loads first: L2-hot) ----
    AFrag a[2][2];                                // [ksub][mi]
#pragma unroll
    for (int ksub = 0; ksub < 2; ++ksub)
#pragma unroll
      for (int mi = 0; mi < 2; ++mi) {
        const uint4* ap = wpk + (((mt0 + mi) * KSTEPS + (s2 * 2 + ksub)) * 32 + lane) * 2;
        a[ksub][mi].u[0] = ap[0]; a[ksub][mi].u[1] = ap[1];
      }
    // prefetch A for the following step
    {
      int spf = (s2 * 2 + 2 < KSTEPS) ? (s2 * 2 + 2) : (KSTEPS - 1);
      __builtin_prefetch(wpk + ((mt0 * KSTEPS + spf) * 32 + lane) * 2, 0, 1);
    }
    // ---- next-tile bilinear sample loads (overlap with WMMAs) ----
    int s2n = (s2 + 1 < NSTEP2) ? (s2 + 1) : (NSTEP2 - 1);
    int kkn = s2n >> 2, cpn = s2n & 3;
    SMeta sm = smeta[kkn * 64 + e];
    int c0 = cpn * 64 + cg * 8;
    v8h v00 = *(const v8h*)(xh + sm.a00 + c0);
    v8h v01 = *(const v8h*)(xh + sm.a01 + c0);
    v8h v10 = *(const v8h*)(xh + sm.a10 + c0);
    v8h v11 = *(const v8h*)(xh + sm.a11 + c0);
    v8h u00 = *(const v8h*)(xh + sm.a00 + c0 + 32);
    v8h u01 = *(const v8h*)(xh + sm.a01 + c0 + 32);
    v8h u10 = *(const v8h*)(xh + sm.a10 + c0 + 32);
    v8h u11 = *(const v8h*)(xh + sm.a11 + c0 + 32);
    // ---- compute current tile: 8 WMMAs per wave ----
    const unsigned int* cb = lcol[s2 & 1];
    int ko = (lane < 16) ? 0 : 4;
#pragma unroll
    for (int ksub = 0; ksub < 2; ++ksub) {
      BFrag bf[2];
#pragma unroll
      for (int ni = 0; ni < 2; ++ni) {
        int pix = waveN * 32 + ni * 16 + (lane & 15);
        int bb = pix * CS64 + ksub * 16;
#pragma unroll
        for (int j = 0; j < 4; ++j) {
          bf[ni].u[j]     = cb[bb + ko + j];
          bf[ni].u[4 + j] = cb[bb + 8 + ko + j];
        }
      }
#pragma unroll
      for (int mi = 0; mi < 2; ++mi)
#pragma unroll
        for (int ni = 0; ni < 2; ++ni)
          acc[mi][ni] = __builtin_amdgcn_wmma_f32_16x16x32_f16(
              false, a[ksub][mi].h, false, bf[ni].h, (short)0, acc[mi][ni], false, false);
    }
    // ---- blend + store next tile into other buffer ----
    _Float16 w00 = (_Float16)sm.w00, w01 = (_Float16)sm.w01;
    _Float16 w10 = (_Float16)sm.w10, w11 = (_Float16)sm.w11;
    HPack r0; r0.h = v00 * w00 + v01 * w01 + v10 * w10 + v11 * w11;
    HPack r1; r1.h = u00 * w00 + u01 * w01 + u10 * w10 + u11 * w11;
    unsigned int* nb = lcol[(s2 + 1) & 1];
    *(uint4*)&nb[e * CS64 + cg * 4]      = r0.u;
    *(uint4*)&nb[e * CS64 + 16 + cg * 4] = r1.u;
  }

  // fused BN + SiLU epilogue
  int nloc = lane & 15;
#pragma unroll
  for (int mi = 0; mi < 2; ++mi) {
    int mloc = (waveM * 2 + mi) * 16 + ((lane >= 16) ? 8 : 0);
#pragma unroll
    for (int ni = 0; ni < 2; ++ni) {
      int n = waveN * 32 + ni * 16 + nloc;
      int p = p0 + n;
      int hw = p & 4095;
#pragma unroll
      for (int r = 0; r < 8; ++r) {
        int ml = mloc + r;
        float v = acc[mi][ni][r] * bnA[ml] + bnB[ml];
        float sv = v / (1.0f + expf(-v));
        out[((b * C2_ + blockIdx.y * 128 + ml) * (H_ * W_)) + hw] = sv;
      }
    }
  }
}

extern "C" void kernel_launch(void* const* d_in, const int* in_sizes, int n_in,
                              void* d_out, int out_size, void* d_ws, size_t ws_size,
                              hipStream_t stream) {
  const float* x        = (const float*)d_in[0];
  const float* w_om     = (const float*)d_in[1];
  const float* b_om     = (const float*)d_in[2];
  const float* w_dcn    = (const float*)d_in[3];
  const float* bn_gamma = (const float*)d_in[4];
  const float* bn_beta  = (const float*)d_in[5];
  const float* bn_mean  = (const float*)d_in[6];
  const float* bn_var   = (const float*)d_in[7];
  float* out = (float*)d_out;

  char* ws = (char*)d_ws;
  _Float16* xh      = (_Float16*)(ws);                       // 8,388,608 B
  _Float16* wdcn_pk = (_Float16*)(ws + 8388608);             // 1,179,648 B
  _Float16* wom_pk  = (_Float16*)(ws + 8388608 + 1179648);   //   147,456 B
  float*    om_ws   = (float*)   (ws + 8388608 + 1179648 + 147456); // 2,097,152 B

  convert_x_kernel<<<16384, 256, 0, stream>>>(x, xh);
  pack_dcn_kernel<<<2304, 256, 0, stream>>>(w_dcn, wdcn_pk);
  pack_om_kernel<<<288, 256, 0, stream>>>(w_om, wom_pk);
  om_conv_kernel<<<NPIX / 64, 256, 0, stream>>>(xh, (const uint4*)wom_pk, b_om, om_ws);
  dcn_main_kernel<<<dim3(NPIX / 64, 2), 256, 0, stream>>>(
      xh, (const uint4*)wdcn_pk, om_ws, bn_gamma, bn_beta, bn_mean, bn_var, out);
}
